// NMTModel_50783693308329
// MI455X (gfx1250) — compile-verified, matrix operand order checked by reading, not run on hardware
//
#include <hip/hip_runtime.h>

// ---------------------------------------------------------------------------
// NMT forward for MI455X (gfx1250, wave32, WMMA + TDM).
// - attention ctx is step-invariant (h-term cancels in softmax) -> computed once
// - all GEMMs: bf16 WMMA, fp32 accumulate; weights bf16-repacked once and
//   L2-resident (proj_W bf16 = 64MB << 192MB L2)
// - decoder GEMMs (M=64): A staged into LDS once per workgroup via the
//   Tensor Data Mover (tensor_load_to_lds + s_wait_tensorcnt), each wave
//   computes a 64x32 strip (8 WMMAs per K-chunk vs 2 global B-fragments)
// ---------------------------------------------------------------------------

typedef __attribute__((ext_vector_type(16))) __bf16 v16bf;
typedef __attribute__((ext_vector_type(8)))  float  v8f;
typedef __attribute__((ext_vector_type(4)))  unsigned int u32x4;
typedef __attribute__((ext_vector_type(8)))  int i32x8;
typedef __attribute__((ext_vector_type(4)))  int i32x4;

#if defined(__has_builtin)
#  if __has_builtin(__builtin_amdgcn_tensor_load_to_lds)
#    define HAVE_TDM 1
#  endif
#endif
#ifndef HAVE_TDM
#  define HAVE_TDM 0
#endif

__device__ __forceinline__ unsigned short f2bf(float f) {
    unsigned u = __float_as_uint(f);
    unsigned r = u + 0x7FFFu + ((u >> 16) & 1u);   // round-to-nearest-even
    return (unsigned short)(r >> 16);
}

union Frag16 { uint4 q[2]; v16bf v; };

// 16x32 bf16 fragment from row-major [rows x ld] (ISA 7.12.2 16-bit A layout):
// lane&15 = row, lane>>4 selects K-halves; two contiguous 16B chunks per lane.
__device__ __forceinline__ v16bf load_frag(const unsigned short* __restrict__ base,
                                           int ld, int row0, int k0) {
    int lane = threadIdx.x & 31;
    const unsigned short* p =
        base + (size_t)(row0 + (lane & 15)) * ld + k0 + ((lane >> 4) << 3);
    Frag16 f;
    f.q[0] = *reinterpret_cast<const uint4*>(p);
    f.q[1] = *reinterpret_cast<const uint4*>(p + 16);
    return f.v;
}

#define WMMA_BF16(acc, a, b) \
    __builtin_amdgcn_wmma_f32_16x16x32_bf16(false, a, false, b, (short)0, acc, false, false)

// ---------------------------------------------------------------------------
// Generic path (used for the M=3200 encoder GEMM): A and W from global.
// C[MxN] = A[MxK](bf16) * W[NxK](bf16)^T + bias. One wave: 64x16 strip.
__global__ __launch_bounds__(256) void wmma_gemm64(
    const unsigned short* __restrict__ A,
    const unsigned short* __restrict__ W,
    const float* __restrict__ bias,
    float* __restrict__ C,
    int M, int N, int K, int act)
{
    int wave   = (int)((blockIdx.x * blockDim.x + threadIdx.x) >> 5);
    int nTiles = N >> 4;
    int total  = (M >> 6) * nTiles;
    if (wave >= total) return;                 // wave-uniform: EXEC stays all-1s
    int ms = wave / nTiles, nt = wave % nTiles;
    int m0 = ms << 6,       n0 = nt << 4;
    int lane = threadIdx.x & 31;
    int col  = n0 + (lane & 15);

    float bv = bias ? bias[col] : 0.0f;
    v8f acc0, acc1, acc2, acc3;
    #pragma unroll
    for (int i = 0; i < 8; ++i) { acc0[i] = bv; acc1[i] = bv; acc2[i] = bv; acc3[i] = bv; }

    const unsigned short* wrow = W + (size_t)col * K + ((lane >> 4) << 3);

    for (int k0 = 0; k0 < K; k0 += 32) {
        if (k0 + 160 <= K)
            __builtin_prefetch(wrow + k0 + 128, 0, 3);   // global_prefetch_b8
        v16bf b  = load_frag(W, K, n0,      k0);
        v16bf a0 = load_frag(A, K, m0,      k0);
        v16bf a1 = load_frag(A, K, m0 + 16, k0);
        v16bf a2 = load_frag(A, K, m0 + 32, k0);
        v16bf a3 = load_frag(A, K, m0 + 48, k0);
        acc0 = WMMA_BF16(acc0, a0, b);
        acc1 = WMMA_BF16(acc1, a1, b);
        acc2 = WMMA_BF16(acc2, a2, b);
        acc3 = WMMA_BF16(acc3, a3, b);
    }

    int rh = (lane >> 4) << 3;
#define STORE_ACC(j, acc)                                                     \
    {   int rb = m0 + 16 * j + rh;                                            \
        _Pragma("unroll")                                                     \
        for (int r = 0; r < 8; ++r) {                                         \
            float v = acc[r];                                                 \
            if (act) v = fmaxf(v, 0.0f);                                      \
            C[(size_t)(rb + r) * N + col] = v;                                \
        }                                                                     \
    }
    STORE_ACC(0, acc0) STORE_ACC(1, acc1) STORE_ACC(2, acc2) STORE_ACC(3, acc3)
#undef STORE_ACC
}

// ---------------------------------------------------------------------------
// Decoder path (M == 64): A (64 x K bf16) staged into LDS once per workgroup
// via the Tensor Data Mover; each wave computes a 64x32 strip (2 N-tiles).
__global__ __launch_bounds__(256) void wmma_gemm64_lds(
    const unsigned short* __restrict__ A,   // 64 x K bf16 row-major
    const unsigned short* __restrict__ W,   // N x K bf16 row-major
    const float* __restrict__ bias,
    float* __restrict__ C,                  // 64 x N f32
    int N, int K, int act)
{
    extern __shared__ unsigned short As[];  // 64*K bf16, dynamic LDS (base off 0)

#if HAVE_TDM
    if (threadIdx.x < 32) {                 // one TDM DMA per workgroup (wave 0)
        unsigned long long ga = (unsigned long long)(size_t)A;
        unsigned w = (unsigned)(K >> 1);    // row length in 4B units
        u32x4 g0;
        g0[0] = 1u;                                           // count=1, user D#
        g0[1] = 0u;                                           // lds_addr = dyn-LDS base
        g0[2] = (unsigned)(ga & 0xFFFFFFFFu);                 // global_addr lo
        g0[3] = (unsigned)((ga >> 32) & 0x01FFFFFFu) | (2u << 30); // addr hi | type=2
        i32x8 g1;
        g1[0] = (int)(2u << 16);                              // data_size = 4B
        g1[1] = (int)((w & 0xFFFFu) << 16);                   // tensor_dim0 lo16
        g1[2] = (int)((w >> 16) & 0xFFFFu) | (64 << 16);      // dim0 hi16 | tensor_dim1=64
        g1[3] = (int)((w & 0xFFFFu) << 16);                   // dim1 hi16=0 | tile_dim0=w
        g1[4] = 64;                                           // tile_dim1=64, tile_dim2=0
        g1[5] = (int)w;                                       // tensor_dim0_stride lo32
        g1[6] = 0;                                            // stride hi | dim1_stride
        g1[7] = 0;
        i32x4 z4 = {0, 0, 0, 0};                              // <=2D: groups 2/3 unused
        i32x8 z8 = {0, 0, 0, 0, 0, 0, 0, 0};                  // unused extra group
        __builtin_amdgcn_tensor_load_to_lds(g0, g1, z4, z4, z8, 0);
        __builtin_amdgcn_s_wait_tensorcnt(0);
    }
#else
    {   // fallback: cooperative b128 copy global -> LDS
        const uint4* src = (const uint4*)A;
        uint4* dst = (uint4*)As;
        int chunks = K << 3;                // 64*K*2/16
        for (int i = threadIdx.x; i < chunks; i += blockDim.x) dst[i] = src[i];
    }
#endif
    __syncthreads();

    int wave   = (int)((blockIdx.x * blockDim.x + threadIdx.x) >> 5);
    int strips = N >> 5;                    // 2 N-tiles per wave
    if (wave >= strips) return;
    int n0   = wave << 5;
    int lane = threadIdx.x & 31;
    int col0 = n0 + (lane & 15);
    int col1 = col0 + 16;

    float bv0 = bias ? bias[col0] : 0.0f;
    float bv1 = bias ? bias[col1] : 0.0f;
    v8f c00, c10, c20, c30, c01, c11, c21, c31;
    #pragma unroll
    for (int i = 0; i < 8; ++i) {
        c00[i] = bv0; c10[i] = bv0; c20[i] = bv0; c30[i] = bv0;
        c01[i] = bv1; c11[i] = bv1; c21[i] = bv1; c31[i] = bv1;
    }

    const unsigned short* wrow = W + (size_t)col0 * K + ((lane >> 4) << 3);

    for (int k0 = 0; k0 < K; k0 += 32) {
        if (k0 + 160 <= K)
            __builtin_prefetch(wrow + k0 + 128, 0, 3);   // global_prefetch_b8
        v16bf b0 = load_frag(W, K, n0,      k0);         // global (L2-resident)
        v16bf b1 = load_frag(W, K, n0 + 16, k0);
        v16bf a0 = load_frag(As, K,  0, k0);             // LDS -> ds_load_b128
        v16bf a1 = load_frag(As, K, 16, k0);
        v16bf a2 = load_frag(As, K, 32, k0);
        v16bf a3 = load_frag(As, K, 48, k0);
        c00 = WMMA_BF16(c00, a0, b0);  c01 = WMMA_BF16(c01, a0, b1);
        c10 = WMMA_BF16(c10, a1, b0);  c11 = WMMA_BF16(c11, a1, b1);
        c20 = WMMA_BF16(c20, a2, b0);  c21 = WMMA_BF16(c21, a2, b1);
        c30 = WMMA_BF16(c30, a3, b0);  c31 = WMMA_BF16(c31, a3, b1);
    }

    int rh = (lane >> 4) << 3;
#define STORE2(j, accA, accB)                                                 \
    {   int rb = 16 * j + rh;                                                 \
        _Pragma("unroll")                                                     \
        for (int r = 0; r < 8; ++r) {                                         \
            float v0 = accA[r], v1 = accB[r];                                 \
            if (act) { v0 = fmaxf(v0, 0.0f); v1 = fmaxf(v1, 0.0f); }          \
            C[(size_t)(rb + r) * N + col0] = v0;                              \
            C[(size_t)(rb + r) * N + col1] = v1;                              \
        }                                                                     \
    }
    STORE2(0, c00, c01) STORE2(1, c10, c11) STORE2(2, c20, c21) STORE2(3, c30, c31)
#undef STORE2
}

// ------------------------- elementwise / setup kernels ----------------------

__global__ void k_f32_to_bf16(const float* __restrict__ s,
                              unsigned short* __restrict__ d, long long n) {
    for (long long i = blockIdx.x * (long long)blockDim.x + threadIdx.x;
         i < n; i += (long long)gridDim.x * blockDim.x)
        d[i] = f2bf(s[i]);
}

// Wcomb[4096 x 2048] = [W_ih | W_hh] in bf16 (K order: [x ; h])
__global__ void k_build_wcomb(const float* __restrict__ wih,
                              const float* __restrict__ whh,
                              unsigned short* __restrict__ d) {
    const long long n = 4096LL * 2048;
    for (long long i = blockIdx.x * (long long)blockDim.x + threadIdx.x;
         i < n; i += (long long)gridDim.x * blockDim.x) {
        int row = (int)(i >> 11), k = (int)(i & 2047);
        float v = (k < 1024) ? wih[(size_t)row * 1024 + k]
                             : whh[(size_t)row * 1024 + (k - 1024)];
        d[i] = f2bf(v);
    }
}

// A_enc[3200 x 1024] = [enc_emb[src] | pos_emb[pos]] in bf16
__global__ void k_build_Aenc(const int* __restrict__ sb, const int* __restrict__ bp,
                             const float* __restrict__ ee, const float* __restrict__ pe,
                             unsigned short* __restrict__ A) {
    const long long n = 3200LL * 1024;
    for (long long i = blockIdx.x * (long long)blockDim.x + threadIdx.x;
         i < n; i += (long long)gridDim.x * blockDim.x) {
        int r = (int)(i >> 10), k = (int)(i & 1023);
        float v = (k < 512) ? ee[(size_t)sb[r] * 512 + k]
                            : pe[(size_t)bp[r] * 512 + (k - 512)];
        A[i] = f2bf(v);
    }
}

// avg over S, output bf16 (A matrix of the h0 GEMM)
__global__ void k_avg(const float* __restrict__ enc, unsigned short* __restrict__ avgbf) {
    int i = blockIdx.x * blockDim.x + threadIdx.x;
    if (i >= 64 * 512) return;
    int b = i >> 9, e = i & 511;
    const float* p = enc + (size_t)b * 50 * 512 + e;
    float s = 0.f;
    for (int t = 0; t < 50; ++t) s += p[t * 512];
    avgbf[i] = f2bf(s * (1.0f / 50.0f));
}

__global__ void k_copy_f32(const float* __restrict__ s, float* __restrict__ d, int n) {
    int i = blockIdx.x * blockDim.x + threadIdx.x;
    if (i < n) d[i] = s[i];
}

// Attention context (constant across decoder steps — h-term cancels in softmax).
__global__ __launch_bounds__(256) void k_ctx(const float* __restrict__ enc,
                                             const int* __restrict__ sb,
                                             const float* __restrict__ attnW,
                                             const float* __restrict__ attnb,
                                             float* __restrict__ ctx) {
    __shared__ float sc[50];
    __shared__ float aw[50];
    __shared__ float ssum;
    int b = blockIdx.x, t = threadIdx.x;
    if (t < 50) {
        const float* row = enc + ((size_t)b * 50 + t) * 512;
        float d = attnb[0];
        for (int k = 0; k < 512; ++k) d += row[k] * attnW[k];
        sc[t] = d;
    }
    __syncthreads();
    if (t < 50) {
        float m = -1e30f;
        for (int s = 0; s < 50; ++s) m = fmaxf(m, sc[s]);
        float e = __expf(sc[t] - m);
        if (sb[b * 50 + t] == 0) e = 0.0f;     // PAD mask, then renormalize
        aw[t] = e;
    }
    __syncthreads();
    if (t == 0) {
        float s = 0.f;
        for (int i = 0; i < 50; ++i) s += aw[i];
        ssum = 1.0f / s;
    }
    __syncthreads();
    for (int e = t; e < 512; e += blockDim.x) {
        float acc = 0.f;
        for (int s = 0; s < 50; ++s) acc += aw[s] * enc[((size_t)b * 50 + s) * 512 + e];
        ctx[(size_t)b * 512 + e] = acc * ssum;
    }
}

// Per-step A matrix: [dec_emb[tok] | ctx | h] -> bf16 (64 x 2048)
__global__ void k_build_Astep(const int* __restrict__ tb, int t,
                              const float* __restrict__ demb,
                              const float* __restrict__ ctx,
                              const float* __restrict__ h,
                              unsigned short* __restrict__ A) {
    int i = blockIdx.x * blockDim.x + threadIdx.x;
    if (i >= 64 * 2048) return;
    int b = i >> 11, k = i & 2047;
    float v;
    if (k < 512)        { int tok = tb[b * 50 + t]; v = demb[(size_t)tok * 512 + k]; }
    else if (k < 1024)  v = ctx[b * 512 + (k - 512)];
    else                v = h[b * 1024 + (k - 1024)];
    A[i] = f2bf(v);
}

// LSTM pointwise; also emits h_new in bf16 as the projection GEMM's A matrix.
__global__ void k_lstm(const float* __restrict__ gates,
                       const float* __restrict__ bih, const float* __restrict__ bhh,
                       float* __restrict__ h, float* __restrict__ c,
                       unsigned short* __restrict__ hbf) {
    int i = blockIdx.x * blockDim.x + threadIdx.x;
    if (i >= 64 * 1024) return;
    int b = i >> 10, j = i & 1023;
    const float* g = gates + (size_t)b * 4096;
    float gi = g[j]        + bih[j]        + bhh[j];
    float gf = g[j + 1024] + bih[j + 1024] + bhh[j + 1024];
    float gg = g[j + 2048] + bih[j + 2048] + bhh[j + 2048];
    float go = g[j + 3072] + bih[j + 3072] + bhh[j + 3072];
    float si = 1.0f / (1.0f + __expf(-gi));
    float sf = 1.0f / (1.0f + __expf(-gf));
    float so = 1.0f / (1.0f + __expf(-go));
    float cn = sf * c[i] + si * tanhf(gg);
    float hn = so * tanhf(cn);
    c[i] = cn;
    h[i] = hn;
    hbf[i] = f2bf(hn);
}

// ---------------------------------------------------------------------------

extern "C" void kernel_launch(void* const* d_in, const int* in_sizes, int n_in,
                              void* d_out, int out_size, void* d_ws, size_t ws_size,
                              hipStream_t stream) {
    (void)in_sizes; (void)n_in; (void)out_size; (void)ws_size;
    const int*   source_batch    = (const int*)  d_in[0];
    const int*   batch_positions = (const int*)  d_in[2];
    const int*   target_batch    = (const int*)  d_in[3];
    const float* enc_emb  = (const float*)d_in[4];
    const float* pos_emb  = (const float*)d_in[5];
    const float* cat_W    = (const float*)d_in[6];
    const float* cat_b    = (const float*)d_in[7];
    const float* scale_W  = (const float*)d_in[8];
    const float* scale_b  = (const float*)d_in[9];
    const float* dec_emb  = (const float*)d_in[10];
    const float* attn_W   = (const float*)d_in[11];
    const float* attn_b   = (const float*)d_in[12];
    const float* W_ih     = (const float*)d_in[13];
    const float* W_hh     = (const float*)d_in[14];
    const float* b_ih     = (const float*)d_in[15];
    const float* b_hh     = (const float*)d_in[16];
    const float* proj_W   = (const float*)d_in[17];
    const float* proj_b   = (const float*)d_in[18];
    float* out = (float*)d_out;

    // -------- workspace carve-out (~100 MB) --------
    char* ws = (char*)d_ws;
    size_t off = 0;
    auto carve = [&](size_t bytes) -> char* {
        char* p = ws + off;
        off += (bytes + 255) & ~(size_t)255;
        return p;
    };
    unsigned short* projW_bf  = (unsigned short*)carve(32000ULL * 1024 * 2); // 64 MB, L2-resident
    unsigned short* wcomb_bf  = (unsigned short*)carve(4096ULL  * 2048 * 2); // 16 MB
    unsigned short* catW_bf   = (unsigned short*)carve(512ULL   * 1024 * 2);
    unsigned short* scaleW_bf = (unsigned short*)carve(1024ULL  * 512  * 2);
    unsigned short* Aenc_bf   = (unsigned short*)carve(3200ULL  * 1024 * 2);
    float*          enc_out   = (float*)         carve(3200ULL  * 512  * 4);
    unsigned short* avg_bf    = (unsigned short*)carve(64ULL    * 512  * 2);
    float*          ctx       = (float*)         carve(64ULL    * 512  * 4);
    float*          h_f32     = (float*)         carve(64ULL    * 1024 * 4);
    float*          c_f32     = (float*)         carve(64ULL    * 1024 * 4);
    float*          gates     = (float*)         carve(64ULL    * 4096 * 4);
    unsigned short* Astep_bf  = (unsigned short*)carve(64ULL    * 2048 * 2);
    unsigned short* h_bf      = (unsigned short*)carve(64ULL    * 1024 * 2);

    const int EB = 256;      // elementwise block
    const int GS = 2048;     // grid-stride blocks

    // -------- one-time weight repacks (fp32 -> bf16) --------
    k_f32_to_bf16<<<GS, EB, 0, stream>>>(proj_W,  projW_bf,  32000LL * 1024);
    k_f32_to_bf16<<<GS, EB, 0, stream>>>(cat_W,   catW_bf,   512LL   * 1024);
    k_f32_to_bf16<<<GS, EB, 0, stream>>>(scale_W, scaleW_bf, 1024LL  * 512);
    k_build_wcomb<<<GS, EB, 0, stream>>>(W_ih, W_hh, wcomb_bf);

    // -------- encoder --------
    k_build_Aenc<<<GS, EB, 0, stream>>>(source_batch, batch_positions,
                                        enc_emb, pos_emb, Aenc_bf);
    // enc_out = Aenc @ cat_W^T + cat_b : M=3200 N=512 K=1024 -> 1600 strips
    wmma_gemm64<<<200, 256, 0, stream>>>(Aenc_bf, catW_bf, cat_b, enc_out,
                                         3200, 512, 1024, 0);
    k_avg<<<128, EB, 0, stream>>>(enc_out, avg_bf);
    // h0 = relu(avg @ scale_W^T + scale_b) : M=64 N=1024 K=512 (LDS A = 64KB)
    wmma_gemm64_lds<<<4, 256, 64 * 512 * 2, stream>>>(avg_bf, scaleW_bf, scale_b,
                                                      h_f32, 1024, 512, 1);
    k_copy_f32<<<256, EB, 0, stream>>>(h_f32, c_f32, 64 * 1024);   // c0 = h0

    // -------- attention context (step-invariant) --------
    k_ctx<<<64, 256, 0, stream>>>(enc_out, source_batch, attn_W, attn_b, ctx);

    // -------- decoder: 49 sequential LSTM steps --------
    for (int t = 0; t < 49; ++t) {
        k_build_Astep<<<512, EB, 0, stream>>>(target_batch, t, dec_emb, ctx,
                                              h_f32, Astep_bf);
        // gates = [x,h] @ [W_ih|W_hh]^T : M=64 N=4096 K=2048 (LDS A = 256KB)
        wmma_gemm64_lds<<<16, 256, 64 * 2048 * 2, stream>>>(Astep_bf, wcomb_bf,
                                                            nullptr, gates,
                                                            4096, 2048, 0);
        k_lstm<<<256, EB, 0, stream>>>(gates, b_ih, b_hh, h_f32, c_f32, h_bf);
        // logits = h_new @ proj_W^T + proj_b : M=64 N=32000 K=1024 (LDS A = 128KB)
        wmma_gemm64_lds<<<125, 256, 64 * 1024 * 2, stream>>>(h_bf, projW_bf, proj_b,
                                                             out + (size_t)t * 64 * 32000,
                                                             32000, 1024, 0);
    }
}
